// mLSTM_38800734552172
// MI455X (gfx1250) — matrix operand order repaired and checked by reading, not compiled, not run on hardware
//
#include <hip/hip_runtime.h>
#include <hip/hip_bf16.h>

// Problem constants (from reference)
#define B_ 8192
#define V_ 50257
#define E_ 1024
#define H_ 2048
#define O_ 2

// K-chunk (bf16 elements) staged through LDS; padded row stride; 32-row A tile.
#define KC 256
#define LDSLD (KC + 8)
#define NSUB (KC / 32)
#define MROWS 32

typedef __bf16 v16bf __attribute__((ext_vector_type(16)));
typedef float  v8f   __attribute__((ext_vector_type(8)));

// float -> bf16 (round to nearest even)
__device__ __forceinline__ unsigned short f2bf(float f) {
    unsigned int u = __float_as_uint(f);
    unsigned int r = u + 0x7FFFu + ((u >> 16) & 1u);
    return (unsigned short)(r >> 16);
}

union FragU {
    uint4 q[2];
    v16bf v;
};

// B-matrix fragment: 32x16 bf16 where B[k][n] = W[n][k], W row-major.
__device__ __forceinline__ v16bf b_frag_at(const unsigned short* base, int koff) {
    FragU f;
    f.q[0] = *(const uint4*)(base + koff);
    f.q[1] = *(const uint4*)(base + koff + 8);
    return f.v;
}

// A fragment from the LDS-staged chunk (row-major MROWS x KC, stride LDSLD).
__device__ __forceinline__ v16bf lds_a_frag(const unsigned short* lA, int sub,
                                            int lane, int mt) {
    int row = (mt << 4) + (lane & 15);
    int kk  = (sub << 5) + ((lane >> 4) << 4);
    const unsigned short* p = lA + row * LDSLD + kk;
    FragU f;
    f.q[0] = *(const uint4*)(p);
    f.q[1] = *(const uint4*)(p + 8);
    return f.v;
}

// Cooperative chunk load: 256 threads fetch a 32 x KC bf16 tile (64B/thread).
__device__ __forceinline__ void load_chunk_global(const unsigned short* X, int ldx,
                                                  int rowBase, int k0, int t,
                                                  uint4& q0, uint4& q1,
                                                  uint4& q2, uint4& q3) {
    int r = t >> 3;            // 0..31
    int c = (t & 7) << 5;      // halves: 0,32,...,224
    const unsigned short* p = X + (long)(rowBase + r) * ldx + k0 + c;
    q0 = *(const uint4*)(p);
    q1 = *(const uint4*)(p + 8);
    q2 = *(const uint4*)(p + 16);
    q3 = *(const uint4*)(p + 24);
}

__device__ __forceinline__ void store_chunk_lds(unsigned short* lA, int t,
                                                uint4 q0, uint4 q1,
                                                uint4 q2, uint4 q3) {
    int r = t >> 3;
    int c = (t & 7) << 5;
    unsigned short* p = lA + r * LDSLD + c;
    *(uint4*)(p)      = q0;
    *(uint4*)(p + 8)  = q1;
    *(uint4*)(p + 16) = q2;
    *(uint4*)(p + 24) = q3;
}

__device__ __forceinline__ v8f wmma_bf16(v16bf a, v16bf b, v8f c) {
    return __builtin_amdgcn_wmma_f32_16x16x32_bf16(
        false, a, false, b, (short)0, c, false, false);
}

__device__ __forceinline__ float sigmoidf_(float x) {
    return 1.0f / (1.0f + __expf(-x));
}

// ---------------------------------------------------------------------------
// Gate-kernel K-phase: 4 weight matrices x 2 M-subtiles (8 accumulators).
// A and B fragments are both double-buffered across substeps so WMMAs only
// take partial counter waits.
// ---------------------------------------------------------------------------
__device__ __forceinline__ void gate_phase(
    const unsigned short* __restrict__ X, int ldx, int nc,
    const unsigned short* __restrict__ W0, const unsigned short* __restrict__ W1,
    const unsigned short* __restrict__ W2, const unsigned short* __restrict__ W3,
    int rowBase, int colBase, int lane, int t,
    unsigned short* lds0, unsigned short* lds1,
    v8f& aF0, v8f& aF1, v8f& aI0, v8f& aI1,
    v8f& aO0, v8f& aO1, v8f& aC0, v8f& aC1) {

    const int n     = colBase + (lane & 15);
    const int laneK = (lane >> 4) << 4;
    const unsigned short* b0 = W0 + (long)n * ldx + laneK;
    const unsigned short* b1 = W1 + (long)n * ldx + laneK;
    const unsigned short* b2 = W2 + (long)n * ldx + laneK;
    const unsigned short* b3 = W3 + (long)n * ldx + laneK;

    uint4 p0, p1, p2, p3;
    load_chunk_global(X, ldx, rowBase, 0, t, p0, p1, p2, p3);
    store_chunk_lds(lds0, t, p0, p1, p2, p3);
    __syncthreads();

    for (int c = 0; c < nc; ++c) {
        unsigned short* cur = (c & 1) ? lds1 : lds0;
        unsigned short* nxt = (c & 1) ? lds0 : lds1;
        const int k0 = c * KC;
        const bool more = (c + 1 < nc);
        if (more) {
            load_chunk_global(X, ldx, rowBase, k0 + KC, t, p0, p1, p2, p3);
            __builtin_prefetch(b0 + k0 + KC, 0, 2);
            __builtin_prefetch(b1 + k0 + KC, 0, 2);
            __builtin_prefetch(b2 + k0 + KC, 0, 2);
            __builtin_prefetch(b3 + k0 + KC, 0, 2);
        }
        // Prime substep 0 fragments.
        v16bf f0 = b_frag_at(b0, k0);
        v16bf f1 = b_frag_at(b1, k0);
        v16bf f2 = b_frag_at(b2, k0);
        v16bf f3 = b_frag_at(b3, k0);
        v16bf a0 = lds_a_frag(cur, 0, lane, 0);
        v16bf a1 = lds_a_frag(cur, 0, lane, 1);
#pragma unroll
        for (int s = 0; s < NSUB; ++s) {
            v16bf f0n = f0, f1n = f1, f2n = f2, f3n = f3;
            v16bf a0n = a0, a1n = a1;
            if (s + 1 < NSUB) {                   // prefetch next substep's frags
                const int kn = k0 + (s + 1) * 32;
                f0n = b_frag_at(b0, kn);
                f1n = b_frag_at(b1, kn);
                f2n = b_frag_at(b2, kn);
                f3n = b_frag_at(b3, kn);
                a0n = lds_a_frag(cur, s + 1, lane, 0);
                a1n = lds_a_frag(cur, s + 1, lane, 1);
            }
            aF0 = wmma_bf16(a0, f0, aF0);  aF1 = wmma_bf16(a1, f0, aF1);
            aI0 = wmma_bf16(a0, f1, aI0);  aI1 = wmma_bf16(a1, f1, aI1);
            aO0 = wmma_bf16(a0, f2, aO0);  aO1 = wmma_bf16(a1, f2, aO1);
            aC0 = wmma_bf16(a0, f3, aC0);  aC1 = wmma_bf16(a1, f3, aC1);
            f0 = f0n; f1 = f1n; f2 = f2n; f3 = f3n;
            a0 = a0n; a1 = a1n;
        }
        if (more) store_chunk_lds(nxt, t, p0, p1, p2, p3);
        __syncthreads();
    }
}

// ---------------------------------------------------------------------------
// m-kernel K-phase: one weight matrix x 2 M-subtiles (2 accumulators).
// ---------------------------------------------------------------------------
__device__ __forceinline__ void mm_phase(
    const unsigned short* __restrict__ X, int ldx, int nc,
    const unsigned short* __restrict__ W,
    int rowBase, int colBase, int lane, int t,
    unsigned short* lds0, unsigned short* lds1,
    v8f& acc0, v8f& acc1) {

    const int laneK = (lane >> 4) << 4;
    const unsigned short* b0 = W + (long)(colBase + (lane & 15)) * ldx + laneK;

    uint4 p0, p1, p2, p3;
    load_chunk_global(X, ldx, rowBase, 0, t, p0, p1, p2, p3);
    store_chunk_lds(lds0, t, p0, p1, p2, p3);
    __syncthreads();

    for (int c = 0; c < nc; ++c) {
        unsigned short* cur = (c & 1) ? lds1 : lds0;
        unsigned short* nxt = (c & 1) ? lds0 : lds1;
        const int k0 = c * KC;
        const bool more = (c + 1 < nc);
        if (more) {
            load_chunk_global(X, ldx, rowBase, k0 + KC, t, p0, p1, p2, p3);
            __builtin_prefetch(b0 + k0 + KC, 0, 2);
        }
        v16bf f0 = b_frag_at(b0, k0);
        v16bf a0 = lds_a_frag(cur, 0, lane, 0);
        v16bf a1 = lds_a_frag(cur, 0, lane, 1);
#pragma unroll
        for (int s = 0; s < NSUB; ++s) {
            v16bf f0n = f0;
            v16bf a0n = a0, a1n = a1;
            if (s + 1 < NSUB) {
                const int kn = k0 + (s + 1) * 32;
                f0n = b_frag_at(b0, kn);
                a0n = lds_a_frag(cur, s + 1, lane, 0);
                a1n = lds_a_frag(cur, s + 1, lane, 1);
            }
            acc0 = wmma_bf16(a0, f0, acc0);
            acc1 = wmma_bf16(a1, f0, acc1);
            f0 = f0n; a0 = a0n; a1 = a1n;
        }
        if (more) store_chunk_lds(nxt, t, p0, p1, p2, p3);
        __syncthreads();
    }
}

// ---------------------------------------------------------------------------
// Kernel 1: embedding gather + cast to bf16.
// ---------------------------------------------------------------------------
__global__ __launch_bounds__(256) void gather_cast_kernel(const int* __restrict__ inp,
                                                          const float* __restrict__ emb,
                                                          unsigned int* __restrict__ xb) {
    long idx = (long)blockIdx.x * blockDim.x + threadIdx.x;   // over B*E/2
    if (idx >= (long)B_ * E_ / 2) return;
    int b  = (int)(idx / (E_ / 2));
    int e2 = (int)(idx % (E_ / 2));
    const float2 v = *(const float2*)(emb + (long)inp[b] * E_ + (long)e2 * 2);
    xb[idx] = (unsigned int)f2bf(v.x) | ((unsigned int)f2bf(v.y) << 16);
}

// ---------------------------------------------------------------------------
// Kernel 2: generic fp32 -> bf16 cast (pairs), grid-strided.
// ---------------------------------------------------------------------------
__global__ __launch_bounds__(256) void cast_bf16_kernel(const float* __restrict__ src,
                                                        unsigned int* __restrict__ dst,
                                                        long n2) {
    long i = (long)blockIdx.x * blockDim.x + threadIdx.x;
    long stride = (long)gridDim.x * blockDim.x;
    for (; i < n2; i += stride) {
        float2 v = *(const float2*)(src + i * 2);
        dst[i] = (unsigned int)f2bf(v.x) | ((unsigned int)f2bf(v.y) << 16);
    }
}

// ---------------------------------------------------------------------------
// Kernel 3: m_t = (x W_mx^T + b_mx) * (h0 W_mh^T + b_mh), bf16 [B,H].
// Block: 8 waves, tile 32 rows x 128 cols; wave w owns cols w*16..w*16+15.
// ---------------------------------------------------------------------------
__global__ __launch_bounds__(256) void m_kernel(const unsigned short* __restrict__ xb,
                                                const unsigned short* __restrict__ h0b,
                                                const unsigned short* __restrict__ Wmx,
                                                const unsigned short* __restrict__ Wmh,
                                                const float* __restrict__ bmx,
                                                const float* __restrict__ bmh,
                                                unsigned short* __restrict__ mb) {
    __shared__ unsigned short lsA[2][MROWS * LDSLD];

    const int t    = threadIdx.x;
    const int lane = t & 31;
    const int wave = t >> 5;
    const int rowBase = blockIdx.x * MROWS;
    const int colBase = blockIdx.y * 128 + wave * 16;

    v8f accx0 = {}, accx1 = {}, acch0 = {}, acch1 = {};

    mm_phase(xb,  E_, E_ / KC, Wmx, rowBase, colBase, lane, t,
             &lsA[0][0], &lsA[1][0], accx0, accx1);
    mm_phase(h0b, H_, H_ / KC, Wmh, rowBase, colBase, lane, t,
             &lsA[0][0], &lsA[1][0], acch0, acch1);

    const int col = colBase + (lane & 15);
    const float bx = bmx[col];
    const float bh = bmh[col];
#pragma unroll
    for (int mt = 0; mt < 2; ++mt) {
        const int rbase = rowBase + (mt << 4) + ((lane >> 4) << 3);
        const v8f& ax = mt ? accx1 : accx0;
        const v8f& ah = mt ? acch1 : acch0;
#pragma unroll
        for (int r = 0; r < 8; ++r) {
            float mval = (ax[r] + bx) * (ah[r] + bh);
            mb[(long)(rbase + r) * H_ + col] = f2bf(mval);
        }
    }
}

// ---------------------------------------------------------------------------
// Kernel 4: all four gates fused + LSTM elementwise -> hx, cx.
// Block tile: 32 rows x 128 cols; wave owns one 16-wide N-subtile x 2 M-subtiles.
// ---------------------------------------------------------------------------
__global__ __launch_bounds__(256) void gate_kernel(
    const unsigned short* __restrict__ xb, const unsigned short* __restrict__ mb,
    const unsigned short* __restrict__ Wfx, const unsigned short* __restrict__ Wfm,
    const unsigned short* __restrict__ Wix, const unsigned short* __restrict__ Wim,
    const unsigned short* __restrict__ Wox, const unsigned short* __restrict__ Wom,
    const unsigned short* __restrict__ Wcx, const unsigned short* __restrict__ Wcm,
    const float* __restrict__ bfx, const float* __restrict__ bfm,
    const float* __restrict__ bix, const float* __restrict__ bim,
    const float* __restrict__ box_, const float* __restrict__ bom,
    const float* __restrict__ bcx, const float* __restrict__ bcm,
    const float* __restrict__ c0,
    float* __restrict__ hx_out, float* __restrict__ cx_out) {

    __shared__ unsigned short lsA[2][MROWS * LDSLD];

    const int t    = threadIdx.x;
    const int lane = t & 31;
    const int wave = t >> 5;
    const int rowBase = blockIdx.x * MROWS;
    const int colBase = blockIdx.y * 128 + wave * 16;

    v8f aF0 = {}, aF1 = {}, aI0 = {}, aI1 = {};
    v8f aO0 = {}, aO1 = {}, aC0 = {}, aC1 = {};

    gate_phase(xb, E_, E_ / KC, Wfx, Wix, Wox, Wcx, rowBase, colBase, lane, t,
               &lsA[0][0], &lsA[1][0], aF0, aF1, aI0, aI1, aO0, aO1, aC0, aC1);
    gate_phase(mb, H_, H_ / KC, Wfm, Wim, Wom, Wcm, rowBase, colBase, lane, t,
               &lsA[0][0], &lsA[1][0], aF0, aF1, aI0, aI1, aO0, aO1, aC0, aC1);

    const int col = colBase + (lane & 15);
    const float bF = bfx[col] + bfm[col];
    const float bI = bix[col] + bim[col];
    const float bO = box_[col] + bom[col];
    const float bC = bcx[col] + bcm[col];

#pragma unroll
    for (int mt = 0; mt < 2; ++mt) {
        const int rbase = rowBase + (mt << 4) + ((lane >> 4) << 3);
        const v8f& vF = mt ? aF1 : aF0;
        const v8f& vI = mt ? aI1 : aI0;
        const v8f& vO = mt ? aO1 : aO0;
        const v8f& vC = mt ? aC1 : aC0;
#pragma unroll
        for (int r = 0; r < 8; ++r) {
            long idx = (long)(rbase + r) * H_ + col;
            float f  = sigmoidf_(vF[r] + bF);
            float i  = sigmoidf_(vI[r] + bI);
            float o  = sigmoidf_(vO[r] + bO);
            float ct = tanhf(vC[r] + bC);
            float cxv = f * c0[idx] + i * ct;
            float hxv = o * tanhf(cxv);
            cx_out[idx] = cxv;
            hx_out[idx] = hxv;
        }
    }
}

// ---------------------------------------------------------------------------
// Kernel 5: out = hx @ W_dec^T + b_dec  (O = 2).
// ---------------------------------------------------------------------------
__global__ __launch_bounds__(256) void decode_kernel(const float* __restrict__ hx,
                                                     const float* __restrict__ Wd,
                                                     const float* __restrict__ bd,
                                                     float* __restrict__ out) {
    __shared__ float s0[256];
    __shared__ float s1[256];
    const int b = blockIdx.x;
    const int t = threadIdx.x;
    float a0 = 0.f, a1 = 0.f;
    for (int h = t; h < H_; h += 256) {
        float v = hx[(long)b * H_ + h];
        a0 += v * Wd[h];
        a1 += v * Wd[H_ + h];
    }
    s0[t] = a0; s1[t] = a1;
    __syncthreads();
    for (int off = 128; off > 0; off >>= 1) {
        if (t < off) { s0[t] += s0[t + off]; s1[t] += s1[t + off]; }
        __syncthreads();
    }
    if (t == 0) {
        out[(long)b * O_ + 0] = s0[0] + bd[0];
        out[(long)b * O_ + 1] = s1[0] + bd[1];
    }
}

// ---------------------------------------------------------------------------
extern "C" void kernel_launch(void* const* d_in, const int* in_sizes, int n_in,
                              void* d_out, int out_size, void* d_ws, size_t ws_size,
                              hipStream_t stream) {
    (void)in_sizes; (void)n_in; (void)out_size; (void)ws_size;

    const int*   inp  = (const int*)d_in[0];
    const float* h_0  = (const float*)d_in[1];
    const float* c_0  = (const float*)d_in[2];
    const float* emb  = (const float*)d_in[3];
    const float* W_mx = (const float*)d_in[4];  const float* b_mx = (const float*)d_in[5];
    const float* W_mh = (const float*)d_in[6];  const float* b_mh = (const float*)d_in[7];
    const float* W_fx = (const float*)d_in[8];  const float* b_fx = (const float*)d_in[9];
    const float* W_fm = (const float*)d_in[10]; const float* b_fm = (const float*)d_in[11];
    const float* W_ix = (const float*)d_in[12]; const float* b_ix = (const float*)d_in[13];
    const float* W_im = (const float*)d_in[14]; const float* b_im = (const float*)d_in[15];
    const float* W_ox = (const float*)d_in[16]; const float* b_ox = (const float*)d_in[17];
    const float* W_om = (const float*)d_in[18]; const float* b_om = (const float*)d_in[19];
    const float* W_cx = (const float*)d_in[20]; const float* b_cx = (const float*)d_in[21];
    const float* W_cm = (const float*)d_in[22]; const float* b_cm = (const float*)d_in[23];
    const float* W_dec = (const float*)d_in[24]; const float* b_dec = (const float*)d_in[25];

    // d_out layout: out [B,O] | hx [B,H] | cx [B,H]  (fp32, reference return order)
    float* out_p = (float*)d_out;
    float* hx_p  = out_p + (size_t)B_ * O_;
    float* cx_p  = hx_p + (size_t)B_ * H_;

    // Workspace layout (bf16 staging)
    char* ws = (char*)d_ws;
    const size_t HE = (size_t)H_ * E_;
    const size_t HH = (size_t)H_ * H_;
    unsigned short* xb  = (unsigned short*)ws;      // B*E
    unsigned short* h0b = xb + (size_t)B_ * E_;     // B*H
    unsigned short* mb  = h0b + (size_t)B_ * H_;    // B*H
    unsigned short* wpool = mb + (size_t)B_ * H_;   // weights (bf16)

    unsigned short* Wb[10];
    const float* wsrc[10] = { W_mx, W_mh, W_fx, W_fm, W_ix, W_im, W_ox, W_om, W_cx, W_cm };
    const size_t  wn[10]  = { HE,   HH,   HE,   HH,   HE,   HH,   HE,   HH,   HE,   HH };
    {
        unsigned short* p = wpool;
        for (int i = 0; i < 10; ++i) { Wb[i] = p; p += wn[i]; }
    }

    // 1) embedding gather + cast
    {
        long pairs = (long)B_ * E_ / 2;
        gather_cast_kernel<<<dim3((unsigned)((pairs + 255) / 256)), dim3(256), 0, stream>>>(
            inp, emb, (unsigned int*)xb);
    }
    // 2) h0 cast
    cast_bf16_kernel<<<dim3(4096), dim3(256), 0, stream>>>(
        h_0, (unsigned int*)h0b, (long)B_ * H_ / 2);
    // 3) weight casts
    for (int i = 0; i < 10; ++i) {
        cast_bf16_kernel<<<dim3(4096), dim3(256), 0, stream>>>(
            wsrc[i], (unsigned int*)Wb[i], (long)(wn[i] / 2));
    }

    // 4) m_t  (tile 32 x 128 per block)
    m_kernel<<<dim3(B_ / MROWS, H_ / 128), dim3(256), 0, stream>>>(
        xb, h0b, Wb[0], Wb[1], b_mx, b_mh, mb);

    // 5) fused gates + LSTM elementwise -> hx, cx  (tile 32 x 128 per block)
    gate_kernel<<<dim3(B_ / MROWS, H_ / 128), dim3(256), 0, stream>>>(
        xb, mb,
        Wb[2], Wb[3], Wb[4], Wb[5], Wb[6], Wb[7], Wb[8], Wb[9],
        b_fx, b_fm, b_ix, b_im, b_ox, b_om, b_cx, b_cm,
        c_0, hx_p, cx_p);

    // 6) decode
    decode_kernel<<<dim3(B_), dim3(256), 0, stream>>>(hx_p, W_dec, b_dec, out_p);
}
